// QKVAttentionWithPosBias_46420006535770
// MI455X (gfx1250) — matrix operand order repaired and checked by reading, not compiled, not run on hardware
//
#include <hip/hip_runtime.h>

#define BATCH 128
#define CDIM  512
#define NH    8
#define HD    64
#define NSEQ  196
#define NPAD  208   // 13*16
#define NT    13
#define PCOLS 224   // 7*32, padded columns for K=32 chunks

typedef __attribute__((ext_vector_type(16))) _Float16 v16h;
typedef __attribute__((ext_vector_type(8)))  _Float16 v8h;
typedef __attribute__((ext_vector_type(8)))  float    v8f;
typedef __attribute__((ext_vector_type(4)))  unsigned v4u;
typedef __attribute__((ext_vector_type(8)))  unsigned v8u;

__device__ __forceinline__ v8f wmma_f16(v16h a, v16h b, v8f c) {
    // D = A*B + C, f16 inputs, f32 accumulate
    return __builtin_amdgcn_wmma_f32_16x16x32_f16(
        /*neg_a=*/false, a, /*neg_b=*/false, b,
        /*c_mod=*/(short)0, c, /*reuse_a=*/false, /*reuse_b=*/false);
}

// ---------------- prep kernels ----------------
__global__ void bias_expand_kernel(const float* __restrict__ ab,
                                   const int* __restrict__ idxs,
                                   float* __restrict__ bias_full) {
    int i = blockIdx.x * blockDim.x + threadIdx.x;
    if (i < NH * NSEQ * NSEQ) {
        int h  = i / (NSEQ * NSEQ);
        int nm = i % (NSEQ * NSEQ);
        bias_full[i] = ab[h * NSEQ + idxs[nm]];
    }
}

__global__ void wconv_kernel(const float* __restrict__ w, _Float16* __restrict__ wh) {
    int i = blockIdx.x * blockDim.x + threadIdx.x;
    if (i < CDIM * CDIM) wh[i] = (_Float16)w[i];
}

// ---------------- attention + gelu ----------------
__global__ __launch_bounds__(128) void attn_gelu_kernel(
    const float* __restrict__ q, const float* __restrict__ k,
    const float* __restrict__ v, const float* __restrict__ bias_full,
    _Float16* __restrict__ xws)
{
    __shared__ __attribute__((aligned(32))) _Float16 KshT[NPAD][HD];     // K^T: [m][d]
    __shared__ __attribute__((aligned(32))) _Float16 Vsh[HD][PCOLS];     // V:   [d][m]
    __shared__ __attribute__((aligned(32))) _Float16 Psh[4][16][PCOLS];  // per-wave P strip

    const int bh = blockIdx.x;
    const int b  = bh >> 3;
    const int h  = bh & 7;
    const int tid  = threadIdx.x;
    const int wave = tid >> 5;
    const int lane = tid & 31;
    const int half = lane >> 4;
    const int l16  = lane & 15;

    const size_t cbase = ((size_t)b * CDIM + h * HD) * NSEQ;
    const float* qb = q + cbase;
    const float* kb = k + cbase;
    const float* vb = v + cbase;

    // Stage K^T (f16) into LDS: KshT[m][d], m>=196 zero-padded.
    for (int t = tid; t < NPAD * HD; t += 128) {
        int d = t & 63;
        int m = t >> 6;
        KshT[m][d] = (m < NSEQ) ? (_Float16)kb[(size_t)d * NSEQ + m] : (_Float16)0.0f;
    }
    // Stage V (f16) into LDS: Vsh[d][m], m>=196 zero-padded.
    for (int t = tid; t < HD * PCOLS; t += 128) {
        int m = t % PCOLS;
        int d = t / PCOLS;
        Vsh[d][m] = (m < NSEQ) ? (_Float16)vb[(size_t)d * NSEQ + m] : (_Float16)0.0f;
    }
    __syncthreads();

    for (int s = wave; s < NT; s += 4) {
        const int nq  = s * 16 + l16;
        const int nqc = (nq < NSEQ) ? nq : (NSEQ - 1);

        // Q A-fragments: lane = row n, K halves: {0..15} / {16..31}
        v16h aq0, aq1;
        #pragma unroll
        for (int e = 0; e < 16; ++e) {
            int dd = e + half * 16;
            aq0[e] = (_Float16)qb[(size_t)dd * NSEQ + nqc];
            aq1[e] = (_Float16)qb[(size_t)(dd + 32) * NSEQ + nqc];
        }

        // S = Q K^T over 13 column tiles (2 WMMAs each, K=64 total)
        float sv[NT][8];
        #pragma unroll
        for (int t = 0; t < NT; ++t) {
            v8f acc = {};
            v16h b0 = *(const v16h*)&KshT[t * 16 + l16][half * 16];
            v16h b1 = *(const v16h*)&KshT[t * 16 + l16][32 + half * 16];
            acc = wmma_f16(aq0, b0, acc);
            acc = wmma_f16(aq1, b1, acc);
            #pragma unroll
            for (int r = 0; r < 8; ++r) sv[t][r] = acc[r];
        }

        // scale + positional bias + column mask (D layout: col=l16, row=half*8+r)
        #pragma unroll
        for (int t = 0; t < NT; ++t) {
            int m = t * 16 + l16;
            bool mok = (m < NSEQ);
            #pragma unroll
            for (int r = 0; r < 8; ++r) {
                int n = s * 16 + half * 8 + r;
                float bia = (mok && n < NSEQ)
                          ? bias_full[((size_t)h * NSEQ + n) * NSEQ + m] : 0.0f;
                float val = sv[t][r] * 0.125f + bia;
                sv[t][r]  = mok ? val : -3.0e38f;
            }
        }

        // row-wise softmax: reduce across the 16-lane half (row lives there)
        float inv_[8];
        #pragma unroll
        for (int r = 0; r < 8; ++r) {
            float mx = -3.0e38f;
            #pragma unroll
            for (int t = 0; t < NT; ++t) mx = fmaxf(mx, sv[t][r]);
            #pragma unroll
            for (int i = 1; i < 16; i <<= 1) mx = fmaxf(mx, __shfl_xor(mx, i, 32));
            float sum = 0.0f;
            #pragma unroll
            for (int t = 0; t < NT; ++t) {
                float e = __expf(sv[t][r] - mx);
                sv[t][r] = e;
                sum += e;
            }
            #pragma unroll
            for (int i = 1; i < 16; i <<= 1) sum += __shfl_xor(sum, i, 32);
            inv_[r] = 1.0f / sum;
        }

        // Store P (f16) to per-wave LDS strip, row-major [16][224], pad cols zeroed
        #pragma unroll
        for (int t = 0; t < NT; ++t)
            #pragma unroll
            for (int r = 0; r < 8; ++r)
                Psh[wave][half * 8 + r][t * 16 + l16] = (_Float16)(sv[t][r] * inv_[r]);
        #pragma unroll
        for (int r = 0; r < 8; ++r)
            Psh[wave][half * 8 + r][NPAD + l16] = (_Float16)0.0f;

        // O = P V^T : 4 d-tiles x 7 K=32 chunks, B-fragments straight from LDS
        #pragma unroll
        for (int dt = 0; dt < 4; ++dt) {
            v8f oacc = {};
            const int d = dt * 16 + l16;
            #pragma unroll
            for (int c = 0; c < 7; ++c) {
                int mb = c * 32;
                v8h a0 = *(const v8h*)&Psh[wave][l16][mb + half * 16];
                v8h a1 = *(const v8h*)&Psh[wave][l16][mb + half * 16 + 8];
                v16h af;
                #pragma unroll
                for (int e = 0; e < 8; ++e) { af[e] = a0[e]; af[e + 8] = a1[e]; }
                v16h bf = *(const v16h*)&Vsh[d][mb + half * 16];
                oacc = wmma_f16(af, bf, oacc);
            }
            // exact GELU, store x as f16 in [b][n(208)][c(512)] layout for the GEMM
            #pragma unroll
            for (int r = 0; r < 8; ++r) {
                int n = s * 16 + half * 8 + r;
                float x = oacc[r];
                float g = 0.5f * x * (1.0f + erff(x * 0.70710678f));
                _Float16 o = (n < NSEQ) ? (_Float16)g : (_Float16)0.0f;
                xws[((size_t)b * NPAD + n) * CDIM + h * HD + d] = o;
            }
        }
    }
}

// ---------------- projection GEMM + BN affine (TDM-staged) ----------------
__global__ __launch_bounds__(512) void proj_bn_kernel(
    const _Float16* __restrict__ Wh, const _Float16* __restrict__ xws,
    const float* __restrict__ gamma, const float* __restrict__ beta,
    const float* __restrict__ rmean, const float* __restrict__ rvar,
    float* __restrict__ out)
{
    __shared__ __attribute__((aligned(32))) _Float16 Xk[NPAD][128];  // [n][i-kb]

    const int b   = blockIdx.x;
    const int mg  = blockIdx.y;
    const int tid  = threadIdx.x;
    const int wave = tid >> 5;
    const int lane = tid & 31;
    const int half = lane >> 4;
    const int l16  = lane & 15;
    const int mt   = mg * 16 + wave;   // output-channel tile (0..31)

    v8f acc[NT];
    #pragma unroll
    for (int nt = 0; nt < NT; ++nt) acc[nt] = (v8f){};

    const uint64_t slab = (uint64_t)(uintptr_t)xws
                        + (uint64_t)b * NPAD * CDIM * sizeof(_Float16);
    const unsigned lds_base = (unsigned)(uintptr_t)&Xk[0][0];

    for (int kb = 0; kb < CDIM; kb += 128) {
        __syncthreads();
        if (wave == 0) {
            // Tensor DMA: 128x208 f16 tile of the [208][512] slab -> LDS, packed rows.
            uint64_t gaddr = slab + (uint64_t)kb * sizeof(_Float16);
            v4u g0;
            g0[0] = 1u;                                   // count=1, user mode
            g0[1] = lds_base;                             // lds_addr
            g0[2] = (unsigned)gaddr;                      // global_addr[31:0]
            g0[3] = (unsigned)((gaddr >> 32) & 0x01FFFFFFu) | 0x80000000u; // addr[56:32] | type=2
            v8u g1;
            g1[0] = 0x00010000u;                          // workgroup_mask=0, data_size=2B
            g1[1] = (unsigned)(CDIM) << 16;               // tensor_dim0 = 512 (lo16)
            g1[2] = (unsigned)(NPAD) << 16;               // dim0 hi=0 | tensor_dim1 = 208 (lo16)
            g1[3] = 128u << 16;                           // dim1 hi=0 | tile_dim0 = 128
            g1[4] = (unsigned)NPAD;                       // tile_dim1 = 208, tile_dim2 = 0
            g1[5] = (unsigned)CDIM;                       // tensor_dim0_stride = 512 (lo32)
            g1[6] = 0u;                                   // stride hi | dim1_stride lo = 0
            g1[7] = 0u;
            asm volatile("tensor_load_to_lds %0, %1" :: "s"(g0), "s"(g1) : "memory");
            __builtin_amdgcn_s_wait_tensorcnt(0);
        }
        __syncthreads();

        #pragma unroll
        for (int kc = 0; kc < 128; kc += 32) {
            // A fragment from Wh[o][i], row-major
            const _Float16* wr =
                Wh + (size_t)(mt * 16 + l16) * CDIM + kb + kc + half * 16;
            v8h w0 = *(const v8h*)wr;
            v8h w1 = *(const v8h*)(wr + 8);
            v16h af;
            #pragma unroll
            for (int e = 0; e < 8; ++e) { af[e] = w0[e]; af[e + 8] = w1[e]; }
            #pragma unroll
            for (int nt = 0; nt < NT; ++nt) {
                v16h bf = *(const v16h*)&Xk[nt * 16 + l16][kc + half * 16];
                acc[nt] = wmma_f16(af, bf, acc[nt]);
            }
        }
    }

    // fused BN affine + store f32 output [b][o][n]
    float invr[8], shr[8];
    #pragma unroll
    for (int r = 0; r < 8; ++r) {
        int o = mt * 16 + half * 8 + r;
        float iv = gamma[o] * rsqrtf(rvar[o] + 1e-5f);
        invr[r] = iv;
        shr[r]  = beta[o] - rmean[o] * iv;
    }
    #pragma unroll
    for (int nt = 0; nt < NT; ++nt) {
        int n = nt * 16 + l16;
        if (n < NSEQ) {
            #pragma unroll
            for (int r = 0; r < 8; ++r) {
                int o = mt * 16 + half * 8 + r;
                out[((size_t)b * CDIM + o) * NSEQ + n] = acc[nt][r] * invr[r] + shr[r];
            }
        }
    }
}

// ---------------- launcher ----------------
extern "C" void kernel_launch(void* const* d_in, const int* in_sizes, int n_in,
                              void* d_out, int out_size, void* d_ws, size_t ws_size,
                              hipStream_t stream) {
    const float* q       = (const float*)d_in[0];
    const float* k       = (const float*)d_in[1];
    const float* v       = (const float*)d_in[2];
    const float* ab      = (const float*)d_in[3];
    const float* proj_w  = (const float*)d_in[4];
    const float* gamma   = (const float*)d_in[5];
    const float* beta    = (const float*)d_in[6];
    const float* rmean   = (const float*)d_in[7];
    const float* rvar    = (const float*)d_in[8];
    const int*   bidx    = (const int*)d_in[9];

    char* ws = (char*)d_ws;
    const size_t BIAS_BYTES = (size_t)NH * NSEQ * NSEQ * sizeof(float);   // 1,229,312
    const size_t WH_BYTES   = (size_t)CDIM * CDIM * sizeof(_Float16);     //   524,288
    float*    bias_full = (float*)ws;
    _Float16* Wh        = (_Float16*)(ws + BIAS_BYTES);
    _Float16* xws       = (_Float16*)(ws + BIAS_BYTES + WH_BYTES);        // 27.3 MB

    bias_expand_kernel<<<(NH * NSEQ * NSEQ + 255) / 256, 256, 0, stream>>>(
        ab, bidx, bias_full);
    wconv_kernel<<<(CDIM * CDIM + 255) / 256, 256, 0, stream>>>(proj_w, Wh);
    attn_gelu_kernel<<<BATCH * NH, 128, 0, stream>>>(q, k, v, bias_full, xws);
    proj_bn_kernel<<<dim3(BATCH, 2), 512, 0, stream>>>(
        Wh, xws, gamma, beta, rmean, rvar, (float*)d_out);
}
